// Encoder_72507637891110
// MI455X (gfx1250) — compile-verified, matrix operand order checked by reading, not run on hardware
//
#include <hip/hip_runtime.h>
#include <hip/hip_bf16.h>

// ---------------------------------------------------------------------------
// 2-layer GCN for MI455X (gfx1250, wave32).
//  - GEMMs: v_wmma_f32_16x16x32_bf16, one wave per 16x64 C-strip (A reuse x4)
//  - edge aggregation: b128 gather + global_atomic_add_f32 (L2-resident:
//    h(25.6MB) + acc(25.6MB) << 192MB L2)
// ---------------------------------------------------------------------------

typedef __bf16 bf16;
typedef __attribute__((ext_vector_type(16))) __bf16 v16bf;
typedef __attribute__((ext_vector_type(8)))  __bf16 v8bf;
typedef __attribute__((ext_vector_type(8)))  float  v8f;

template <int W> struct vecf;
template <> struct vecf<2> { typedef __attribute__((ext_vector_type(2))) float T; };
template <> struct vecf<4> { typedef __attribute__((ext_vector_type(4))) float T; };

__device__ __forceinline__ bf16 f2bf(float f) {
  union { float f; unsigned u; } in; in.f = f;
  unsigned u = in.u;
  unsigned r = u + 0x7fffu + ((u >> 16) & 1u);   // round-to-nearest-even
  union { unsigned short s; bf16 b; } out;
  out.s = (unsigned short)(r >> 16);
  return out.b;
}

// ---------------------------- degree / norm --------------------------------

__global__ __launch_bounds__(256) void k_deg_init(float* deg, int n) {
  int i = blockIdx.x * 256 + threadIdx.x;
  if (i < n) deg[i] = 1.0f;                      // self loop contributes 1
}

__global__ __launch_bounds__(256) void k_deg_count(const int* __restrict__ dst,
                                                   float* deg, int E) {
  int e = blockIdx.x * 256 + threadIdx.x;
  if (e < E) atomicAdd(&deg[dst[e]], 1.0f);
}

__global__ __launch_bounds__(256) void k_deg_rsqrt(float* deg, int n) {
  int i = blockIdx.x * 256 + threadIdx.x;
  if (i < n) deg[i] = rsqrtf(deg[i]);            // deg >= 1 always
}

// ---------------------------- fp32 -> bf16 ---------------------------------

__global__ __launch_bounds__(256) void k_cvt_bf16(const float* __restrict__ in,
                                                  bf16* __restrict__ out, int n) {
  int i = blockIdx.x * 256 + threadIdx.x;
  if (i < n) out[i] = f2bf(in[i]);
}

__global__ __launch_bounds__(256) void k_relu_cvt_bf16(const float* __restrict__ in,
                                                       bf16* __restrict__ out, int n) {
  int i = blockIdx.x * 256 + threadIdx.x;
  if (i < n) out[i] = f2bf(fmaxf(in[i], 0.0f));
}

// Wt[n*K + k] = W[k*cols + n]  (transpose so B-fragment loads are contiguous in K)
__global__ __launch_bounds__(256) void k_transpose_bf16(const float* __restrict__ W,
                                                        bf16* __restrict__ Wt,
                                                        int K, int cols) {
  int idx = blockIdx.x * 256 + threadIdx.x;
  if (idx >= K * cols) return;
  int n = idx / K;
  int k = idx % K;
  Wt[idx] = f2bf(W[(size_t)k * cols + n]);
}

// ---------------------------- WMMA GEMM ------------------------------------
// C[M x NCOLS] = A[M x 128] * B[128 x NCOLS], Bt = B^T [NCOLS x 128].
// One wave computes a 16x64 strip of C: 4 accumulators share one A fragment
// per K-step (4x less A traffic, 4 back-to-back WMMAs per A load).
// A fragment (ISA 7.12.2, 16-bit 16x32): lane<16 -> row=lane, K {0..7,16..23};
// lane>=16 -> row=lane-16, K {8..15,24..31}.  B symmetric on columns.

template <int NCOLS>  // 128 or 64
__global__ __launch_bounds__(256) void k_gemm_wmma_bf16(
    const bf16* __restrict__ A,   // [M x 128]
    const bf16* __restrict__ Bt,  // [NCOLS x 128]
    float* __restrict__ C,        // [M x NCOLS]
    int M)                        // multiple of 16
{
  constexpr int K = 128;
  constexpr int NT = 4;                 // 4 x 16-wide n-subtiles = 64 cols/wave
  constexpr int chunksN = NCOLS / 64;   // 2 for 128, 1 for 64
  const int wave = threadIdx.x >> 5;
  const int lane = threadIdx.x & 31;
  const int tile = blockIdx.x * 8 + wave;
  const int numTiles = (M >> 4) * chunksN;
  if (tile >= numTiles) return;         // wave-uniform: EXEC all-1s at WMMA
  const int tm = (tile / chunksN) << 4;
  const int tn = (tile % chunksN) << 6;
  const int rc = lane & 15;             // row (A) / col (B, C)
  const int hi = lane >> 4;

  const bf16* aRow  = A  + (size_t)(tm + rc) * K + hi * 8;
  const bf16* bBase = Bt + (size_t)(tn + rc) * K + hi * 8;

  v8f acc[NT] = {};
#pragma unroll
  for (int k0 = 0; k0 < K; k0 += 32) {
    union { v16bf v; v8bf h[2]; } a;
    a.h[0] = *(const v8bf*)(aRow + k0);          // K {0..7}+hi*8
    a.h[1] = *(const v8bf*)(aRow + k0 + 16);     // K {16..23}+hi*8
#pragma unroll
    for (int t = 0; t < NT; ++t) {
      union { v16bf v; v8bf h[2]; } b;
      const bf16* bp = bBase + (size_t)t * 16 * K + k0;
      b.h[0] = *(const v8bf*)(bp);
      b.h[1] = *(const v8bf*)(bp + 16);
      acc[t] = __builtin_amdgcn_wmma_f32_16x16x32_bf16(
          /*neg_a=*/false, a.v, /*neg_b=*/false, b.v,
          /*c_mod=*/(short)0, acc[t], /*reuse_a=*/false, /*reuse_b=*/false);
    }
  }

  // C/D layout: VGPR j -> row (j + hi*8), col = lane&15 (+16 per subtile)
  float* cBase = C + (size_t)tm * NCOLS + tn;
#pragma unroll
  for (int t = 0; t < NT; ++t)
#pragma unroll
    for (int j = 0; j < 8; ++j)
      cBase[(size_t)(j + hi * 8) * NCOLS + t * 16 + rc] = acc[t][j];
}

// ---------------------------- aggregation ----------------------------------

// acc[i][c] = bias[c] + dinv[i]^2 * H[i][c]   (self-loop + bias fused)
template <int NCOLS>
__global__ __launch_bounds__(256) void k_self_init(const float* __restrict__ H,
                                                   const float* __restrict__ bias,
                                                   const float* __restrict__ dinv,
                                                   float* __restrict__ acc, int n) {
  int idx = blockIdx.x * 256 + threadIdx.x;
  if (idx >= n * NCOLS) return;
  int i = idx / NCOLS;
  int c = idx - i * NCOLS;
  float di = dinv[i];
  acc[idx] = bias[c] + di * di * H[idx];
}

// One wave per edge: acc[dst] += H[src] * (dinv[src]*dinv[dst]).
// Lane owns NCOLS/32 contiguous channels: one b64/b128 gather + contiguous atomics.
template <int NCOLS>
__global__ __launch_bounds__(256) void k_edge_scatter(const int* __restrict__ src,
                                                      const int* __restrict__ dst,
                                                      const float* __restrict__ dinv,
                                                      const float* __restrict__ H,
                                                      float* __restrict__ acc, int E) {
  constexpr int VW = NCOLS / 32;        // 4 (layer1) or 2 (layer2)
  typedef typename vecf<VW>::T VecT;
  int wave = (blockIdx.x * 256 + threadIdx.x) >> 5;
  int lane = threadIdx.x & 31;
  if (wave >= E) return;
  int s = src[wave];
  int d = dst[wave];
  float norm = dinv[s] * dinv[d];
  const float* hrow = H + (size_t)s * NCOLS + lane * VW;
  float* arow = acc + (size_t)d * NCOLS + lane * VW;
  VecT hv = *(const VecT*)hrow;
#pragma unroll
  for (int j = 0; j < VW; ++j)
    atomicAdd(&arow[j], hv[j] * norm);
}

// ---------------------------------------------------------------------------

extern "C" void kernel_launch(void* const* d_in, const int* in_sizes, int n_in,
                              void* d_out, int out_size, void* d_ws, size_t ws_size,
                              hipStream_t stream) {
  (void)n_in; (void)out_size; (void)ws_size;
  const float* x  = (const float*)d_in[0];   // [N,128]
  const int*   ei = (const int*)  d_in[1];   // [2,E]
  const float* W1 = (const float*)d_in[2];   // [128,128]
  const float* b1 = (const float*)d_in[3];   // [128]
  const float* W2 = (const float*)d_in[4];   // [128,64]
  const float* b2 = (const float*)d_in[5];   // [64]
  float* out = (float*)d_out;                // [N,64]

  const int IN = 128, HID = 128, OUT = 64;
  const int N = in_sizes[0] / IN;            // 50000
  const int E = in_sizes[1] / 2;             // 800000
  const int M16 = ((N + 15) / 16) * 16;      // 50000 = 3125*16 exactly
  const int* src = ei;
  const int* dst = ei + E;

  // workspace sub-allocation (hipMalloc base is 256B aligned)
  char* ws = (char*)d_ws;
  size_t off = 0;
  auto alloc = [&](size_t bytes) -> void* {
    void* p = ws + off;
    off = (off + bytes + 255) & ~(size_t)255;
    return p;
  };
  float* dinv = (float*)alloc((size_t)N * 4);            //   0.2 MB
  bf16*  Xb   = (bf16*) alloc((size_t)M16 * IN * 2);     //  12.8 MB
  bf16*  W1t  = (bf16*) alloc((size_t)HID * IN * 2);     //  32 KB  [HID x IN]
  bf16*  W2t  = (bf16*) alloc((size_t)OUT * HID * 2);    //  16 KB  [OUT x HID]
  float* h1   = (float*)alloc((size_t)M16 * HID * 4);    //  25.6 MB
  float* acc1 = (float*)alloc((size_t)N * HID * 4);      //  25.6 MB
  bf16*  h1b  = (bf16*) alloc((size_t)M16 * HID * 2);    //  12.8 MB
  float* h2   = (float*)alloc((size_t)M16 * OUT * 4);    //  12.8 MB

  // --- degrees & symmetric norm ---
  k_deg_init <<<(N + 255) / 256, 256, 0, stream>>>(dinv, N);
  k_deg_count<<<(E + 255) / 256, 256, 0, stream>>>(dst, dinv, E);
  k_deg_rsqrt<<<(N + 255) / 256, 256, 0, stream>>>(dinv, N);

  // --- bf16 conversions ---
  k_cvt_bf16      <<<(N * IN + 255) / 256, 256, 0, stream>>>(x, Xb, N * IN);
  k_transpose_bf16<<<(IN * HID + 255) / 256, 256, 0, stream>>>(W1, W1t, IN, HID);
  k_transpose_bf16<<<(HID * OUT + 255) / 256, 256, 0, stream>>>(W2, W2t, HID, OUT);

  // --- layer 1: h1 = Xb @ W1, acc1 = b1 + dinv^2*h1 + edge scatter, relu->bf16
  {
    int tiles = (M16 / 16) * (HID / 64);
    k_gemm_wmma_bf16<HID><<<(tiles + 7) / 8, 256, 0, stream>>>(Xb, W1t, h1, M16);
    k_self_init<HID><<<(N * HID + 255) / 256, 256, 0, stream>>>(h1, b1, dinv, acc1, N);
    k_edge_scatter<HID><<<(E + 7) / 8, 256, 0, stream>>>(src, dst, dinv, h1, acc1, E);
    k_relu_cvt_bf16<<<(N * HID + 255) / 256, 256, 0, stream>>>(acc1, h1b, N * HID);
  }

  // --- layer 2: h2 = h1b @ W2, out = b2 + dinv^2*h2 + edge scatter ---
  {
    int tiles = (M16 / 16) * (OUT / 64);
    k_gemm_wmma_bf16<OUT><<<(tiles + 7) / 8, 256, 0, stream>>>(h1b, W2t, h2, M16);
    k_self_init<OUT><<<(N * OUT + 255) / 256, 256, 0, stream>>>(h2, b2, dinv, out, N);
    k_edge_scatter<OUT><<<(E + 7) / 8, 256, 0, stream>>>(src, dst, dinv, h2, out, E);
  }
}
